// DGLRGATModel_11081015623725
// MI455X (gfx1250) — compile-verified, matrix operand order checked by reading, not run on hardware
//
#include <hip/hip_runtime.h>
#include <stdint.h>

typedef unsigned short u16;
typedef unsigned int   u32;
typedef __attribute__((ext_vector_type(16))) __bf16 v16bf;
typedef __attribute__((ext_vector_type(8)))  float  v8f;
typedef __attribute__((ext_vector_type(4)))  u32    u32x4;

#define NEG_SLOPE 0.2f

// ---------- device helpers ----------
static __device__ __forceinline__ u16 f2bf(float f) {
  u32 b = __float_as_uint(f);
  b += 0x7fffu + ((b >> 16) & 1u);          // round-to-nearest-even bf16
  return (u16)(b >> 16);
}
static __device__ __forceinline__ u32 ordf(float f) {   // order-preserving uint encoding
  u32 b = __float_as_uint(f);
  return (b & 0x80000000u) ? ~b : (b | 0x80000000u);
}
static __device__ __forceinline__ float unordf(u32 o) {
  u32 b = (o & 0x80000000u) ? (o & 0x7fffffffu) : ~o;
  return __uint_as_float(b);
}

// ---------- conversion / init kernels ----------
__global__ void k_f32_to_bf16(const float* __restrict__ in, u16* __restrict__ out, long n) {
  long t = (long)blockIdx.x * blockDim.x + threadIdx.x;
  if (t < n) out[t] = f2bf(in[t]);
}

// W[K][NC] (f32, row-major) -> Wt[NC][K] (bf16) so WMMA B-fragment loads are contiguous
__global__ void k_w_transpose_bf16(const float* __restrict__ W, u16* __restrict__ Wt,
                                   int K, int NC) {
  int t = blockIdx.x * blockDim.x + threadIdx.x;
  if (t >= K * NC) return;
  int k = t / NC, n = t - k * NC;
  Wt[(size_t)n * K + k] = f2bf(W[t]);
}

__global__ void k_fill_f32(float* __restrict__ p, float v, long n) {
  long t = (long)blockIdx.x * blockDim.x + threadIdx.x;
  if (t < n) p[t] = v;
}
__global__ void k_fill_u32(u32* __restrict__ p, u32 v, long n) {
  long t = (long)blockIdx.x * blockDim.x + threadIdx.x;
  if (t < n) p[t] = v;
}
// acc[n, j] = b0[j] (+ b1[j])   for j in [0, HD)
__global__ void k_bias_init(float* __restrict__ acc, const float* __restrict__ b0,
                            const float* __restrict__ b1, int N, int HD) {
  long t = (long)blockIdx.x * blockDim.x + threadIdx.x;
  if (t >= (long)N * HD) return;
  int j = (int)(t % HD);
  float v = b0[j];
  if (b1) v += b1[j];
  acc[t] = v;
}

// ---------- WMMA GEMM: C[M,NC] = A[M,K] (bf16) x Bt[NC,K]^T (bf16), f32 accumulate ----------
// One wave computes one 16x16 tile. K multiple of 32, M multiple of 16, NC multiple of 16.
__global__ void k_gemm_bf16(const u16* __restrict__ A, const u16* __restrict__ Bt,
                            float* __restrict__ C, int M, int K, int NC) {
  int wave = (int)((blockIdx.x * blockDim.x + threadIdx.x) >> 5);
  int lane = (int)(threadIdx.x & 31u);
  int tilesN = NC >> 4;
  int tm = wave / tilesN;
  int tn = wave - tm * tilesN;
  if (tm >= (M >> 4)) return;
  int r16   = lane & 15;
  int khalf = (lane >> 4) << 3;            // 0 or 8 (ISA 16-bit fragment K layout)
  const u16* arow = A  + (size_t)(tm * 16 + r16) * K;
  const u16* brow = Bt + (size_t)(tn * 16 + r16) * K;
  v8f acc = {0.f, 0.f, 0.f, 0.f, 0.f, 0.f, 0.f, 0.f};
  for (int kb = 0; kb < K; kb += 32) {
    union { u32x4 q[2]; v16bf v; } a, b;
    a.q[0] = *(const u32x4*)(arow + kb + khalf);
    a.q[1] = *(const u32x4*)(arow + kb + 16 + khalf);
    b.q[0] = *(const u32x4*)(brow + kb + khalf);
    b.q[1] = *(const u32x4*)(brow + kb + 16 + khalf);
    if (kb + 32 < K) {                     // global_prefetch_b8 next K-block
      __builtin_prefetch(arow + kb + 32, 0, 1);
      __builtin_prefetch(brow + kb + 32, 0, 1);
    }
    acc = __builtin_amdgcn_wmma_f32_16x16x32_bf16(
        /*neg_a=*/false, a.v, /*neg_b=*/false, b.v,
        /*c_mod=*/(short)0, acc, /*reuse_a=*/false, /*reuse_b=*/false);
  }
  // C/D layout: lane -> col = lane%16, VGPR r -> row = r + 8*(lane/16)
  int row0 = tm * 16 + ((lane >> 4) << 3);
  int col  = tn * 16 + r16;
  float* crow = C + (size_t)row0 * NC + col;
#pragma unroll
  for (int r = 0; r < 8; ++r) crow[(size_t)r * NC] = acc[r];
}

// ---------- attention logits: out[n,h] = sum_d f[n,h*D+d] * avec[h*D+d] ----------
__global__ void k_logits(const float* __restrict__ f, const float* __restrict__ avec,
                         float* __restrict__ out, int N, int H, int D) {
  int t = blockIdx.x * blockDim.x + threadIdx.x;
  if (t >= N * H) return;
  int n = t / H, h = t - n * H;
  const float* fr = f + ((size_t)n * H + h) * D;
  const float* ar = avec + h * D;
  float s = 0.f;
#pragma unroll 8
  for (int d = 0; d < D; ++d) s += fr[d] * ar[d];
  out[t] = s;
}

// ---------- edge softmax phase 1: logit + leaky relu + segment max ----------
__global__ void k_edge_logit_max(const float* __restrict__ el, const float* __restrict__ er,
                                 const int* __restrict__ u, const int* __restrict__ v,
                                 float* __restrict__ exbuf, u32* __restrict__ mbuf,
                                 int E, int H) {
  int t = blockIdx.x * blockDim.x + threadIdx.x;
  if (t >= E * H) return;
  int e = t / H, h = t - e * H;
  float x = el[u[e] * H + h] + er[v[e] * H + h];
  x = (x > 0.f) ? x : NEG_SLOPE * x;
  exbuf[t] = x;
  atomicMax(&mbuf[v[e] * H + h], ordf(x));
}

// ---------- edge softmax phase 2: exp + segment sum ----------
__global__ void k_edge_exp(float* __restrict__ exbuf, const u32* __restrict__ mbuf,
                           const int* __restrict__ v, float* __restrict__ sbuf,
                           int E, int H) {
  int t = blockIdx.x * blockDim.x + threadIdx.x;
  if (t >= E * H) return;
  int e = t / H, h = t - e * H;
  int vh = v[e] * H + h;
  float m = unordf(mbuf[vh]);
  if (m < -3.0e38f) m = 0.f;               // guard empty segments (-inf)
  float ex = __expf(exbuf[t] - m);
  exbuf[t] = ex;
  atomicAdd(&sbuf[vh], ex);
}

// ---------- edge scatter: acc[v,h,:] += (ex/s[v,h]) * fs[u,h,:]  (wave per (e,h), lane=d) ----------
__global__ void k_edge_scatter(const float* __restrict__ exbuf, const float* __restrict__ sbuf,
                               const float* __restrict__ fs,
                               const int* __restrict__ u, const int* __restrict__ v,
                               float* __restrict__ acc, int E, int H, int D) {
  int gw   = (int)(((long)blockIdx.x * blockDim.x + threadIdx.x) >> 5);
  int lane = (int)(threadIdx.x & 31u);
  if (gw >= E * H) return;
  int e = gw / H, h = gw - e * H;
  int ue = u[e], ve = v[e];
  float a = exbuf[(size_t)e * H + h] / sbuf[ve * H + h];
  float val = a * fs[((size_t)ue * H + h) * D + lane];
  atomicAdd(&acc[((size_t)ve * H + h) * D + lane], val);
}

// ---------- head mean (+optional relu), optional f32 and bf16 outputs ----------
__global__ void k_head_mean(const float* __restrict__ acc, float* __restrict__ outf,
                            u16* __restrict__ outbf, int N, int H, int D, int do_relu) {
  int t = blockIdx.x * blockDim.x + threadIdx.x;
  if (t >= N * D) return;
  int n = t / D, d = t - n * D;
  float s = 0.f;
#pragma unroll
  for (int h = 0; h < 8; ++h) s += acc[((size_t)n * H + h) * D + d];
  s *= (1.0f / (float)H);
  if (do_relu && s < 0.f) s = 0.f;
  if (outf)  outf[t]  = s;
  if (outbf) outbf[t] = f2bf(s);
}

// ---------- host ----------
static inline unsigned blks(long n) { return (unsigned)((n + 255) / 256); }

extern "C" void kernel_launch(void* const* d_in, const int* in_sizes, int n_in,
                              void* d_out, int out_size, void* d_ws, size_t ws_size,
                              hipStream_t stream) {
  const int Fin = 256, H = 8, D = 32, HD = 256;   // H*D == 256
  const float* xA  = (const float*)d_in[0];
  const float* xB  = (const float*)d_in[1];
  const float* W1  = (const float*)d_in[2];   // [3, 256, 256]
  const float* al1 = (const float*)d_in[3];   // [3, 8, 32]
  const float* ar1 = (const float*)d_in[4];
  const float* b1  = (const float*)d_in[5];   // [3, 256]
  const float* W2  = (const float*)d_in[6];   // [3, 32, 256]
  const float* al2 = (const float*)d_in[7];
  const float* ar2 = (const float*)d_in[8];
  const float* b2  = (const float*)d_in[9];
  const int* U[3] = { (const int*)d_in[10], (const int*)d_in[12], (const int*)d_in[14] };
  const int* V[3] = { (const int*)d_in[11], (const int*)d_in[13], (const int*)d_in[15] };
  const int N = in_sizes[0] / Fin;
  const int E = in_sizes[10];

  // bump allocator over workspace, 256B aligned
  char* wp = (char*)d_ws;
  auto bump = [&](size_t bytes) -> char* {
    char* r = wp;
    wp += (bytes + 255) & ~(size_t)255;
    return r;
  };
  u16*   xAbf  = (u16*)bump((size_t)N * Fin * 2);
  u16*   xBbf  = (u16*)bump((size_t)N * Fin * 2);
  u16*   Wt1   = (u16*)bump((size_t)3 * HD * Fin * 2);   // [e][NC=256][K=256]
  u16*   Wt2   = (u16*)bump((size_t)3 * HD * D * 2);     // [e][NC=256][K=32]
  float* fsb   = (float*)bump((size_t)N * HD * 4);
  float* fdb   = (float*)bump((size_t)N * HD * 4);
  float* el    = (float*)bump((size_t)N * H * 4);
  float* er    = (float*)bump((size_t)N * H * 4);
  u32*   mbuf  = (u32*)bump((size_t)N * H * 4);
  float* sbuf  = (float*)bump((size_t)N * H * 4);
  float* exbuf = (float*)bump((size_t)E * H * 4);
  float* accA  = (float*)bump((size_t)N * HD * 4);
  float* accB  = (float*)bump((size_t)N * HD * 4);
  u16*   hAbf  = (u16*)bump((size_t)N * D * 2);
  u16*   hBbf  = (u16*)bump((size_t)N * D * 2);
  (void)ws_size; (void)n_in; (void)out_size;

  const u32 ORD_NEG_INF = 0x007FFFFFu;   // ordf(-inf)
  const long EH = (long)E * H;

  // one GAT conv: project src/dst, edge softmax over v, weighted scatter into acc
  auto run_conv = [&](const u16* srcbf, const u16* dstbf, int Kin, const u16* Wt,
                      const float* al, const float* ar,
                      const int* u, const int* v, float* acc) {
    long gemm_threads = (long)(N / 16) * (HD / 16) * 32;
    k_gemm_bf16<<<blks(gemm_threads), 256, 0, stream>>>(srcbf, Wt, fsb, N, Kin, HD);
    const float* fdp = fsb;
    if (dstbf != srcbf) {
      k_gemm_bf16<<<blks(gemm_threads), 256, 0, stream>>>(dstbf, Wt, fdb, N, Kin, HD);
      fdp = fdb;
    }
    k_logits<<<blks((long)N * H), 256, 0, stream>>>(fsb, al, el, N, H, D);
    k_logits<<<blks((long)N * H), 256, 0, stream>>>(fdp, ar, er, N, H, D);
    k_fill_u32<<<blks((long)N * H), 256, 0, stream>>>(mbuf, ORD_NEG_INF, (long)N * H);
    k_fill_f32<<<blks((long)N * H), 256, 0, stream>>>(sbuf, 0.f, (long)N * H);
    k_edge_logit_max<<<blks(EH), 256, 0, stream>>>(el, er, u, v, exbuf, mbuf, E, H);
    k_edge_exp<<<blks(EH), 256, 0, stream>>>(exbuf, mbuf, v, sbuf, E, H);
    k_edge_scatter<<<blks(EH * 32), 256, 0, stream>>>(exbuf, sbuf, fsb, u, v, acc, E, H, D);
  };

  // ---- phase 0: precision conversion + weight transpose ----
  k_f32_to_bf16<<<blks((long)N * Fin), 256, 0, stream>>>(xA, xAbf, (long)N * Fin);
  k_f32_to_bf16<<<blks((long)N * Fin), 256, 0, stream>>>(xB, xBbf, (long)N * Fin);
  for (int e = 0; e < 3; ++e) {
    k_w_transpose_bf16<<<blks((long)Fin * HD), 256, 0, stream>>>(
        W1 + (size_t)e * Fin * HD, Wt1 + (size_t)e * HD * Fin, Fin, HD);
    k_w_transpose_bf16<<<blks((long)D * HD), 256, 0, stream>>>(
        W2 + (size_t)e * D * HD, Wt2 + (size_t)e * HD * D, D, HD);
  }

  // ---- layer 1 ----
  k_bias_init<<<blks((long)N * HD), 256, 0, stream>>>(accB, b1 + 0 * HD, nullptr, N, HD);
  k_bias_init<<<blks((long)N * HD), 256, 0, stream>>>(accA, b1 + 1 * HD, b1 + 2 * HD, N, HD);
  // etype0: A -> B
  run_conv(xAbf, xBbf, Fin, Wt1 + (size_t)0 * HD * Fin, al1 + 0 * HD, ar1 + 0 * HD,
           U[0], V[0], accB);
  // etype1: B -> A
  run_conv(xBbf, xAbf, Fin, Wt1 + (size_t)1 * HD * Fin, al1 + 1 * HD, ar1 + 1 * HD,
           U[1], V[1], accA);
  // etype2: A -> A (single projection)
  run_conv(xAbf, xAbf, Fin, Wt1 + (size_t)2 * HD * Fin, al1 + 2 * HD, ar1 + 2 * HD,
           U[2], V[2], accA);

  // ---- head mean + ReLU -> bf16 features for layer 2 ----
  k_head_mean<<<blks((long)N * D), 256, 0, stream>>>(accA, nullptr, hAbf, N, H, D, 1);
  k_head_mean<<<blks((long)N * D), 256, 0, stream>>>(accB, nullptr, hBbf, N, H, D, 1);

  // ---- layer 2 (reuse accumulators) ----
  k_bias_init<<<blks((long)N * HD), 256, 0, stream>>>(accB, b2 + 0 * HD, nullptr, N, HD);
  k_bias_init<<<blks((long)N * HD), 256, 0, stream>>>(accA, b2 + 1 * HD, b2 + 2 * HD, N, HD);
  run_conv(hAbf, hBbf, D, Wt2 + (size_t)0 * HD * D, al2 + 0 * HD, ar2 + 0 * HD,
           U[0], V[0], accB);
  run_conv(hBbf, hAbf, D, Wt2 + (size_t)1 * HD * D, al2 + 1 * HD, ar2 + 1 * HD,
           U[1], V[1], accA);
  run_conv(hAbf, hAbf, D, Wt2 + (size_t)2 * HD * D, al2 + 2 * HD, ar2 + 2 * HD,
           U[2], V[2], accA);

  // ---- final head mean -> stacked [2, N, 32] f32 output ----
  float* out = (float*)d_out;
  k_head_mean<<<blks((long)N * D), 256, 0, stream>>>(accA, out, nullptr, N, H, D, 0);
  k_head_mean<<<blks((long)N * D), 256, 0, stream>>>(accB, out + (size_t)N * D, nullptr, N, H, D, 0);
}
